// FusionMamba_24223615549515
// MI455X (gfx1250) — matrix operand order repaired and checked by reading, not compiled, not run on hardware
//
#include <hip/hip_runtime.h>
#include <hip/hip_fp16.h>
#include <math.h>

// ---------------- problem constants ----------------
#define DIMC   64
#define BATCH  8
#define LSEQ   16384
#define NSTATE 8
#define DTRANK 4
#define DCONV  4
#define ROWS   (BATCH * LSEQ)        // 131072
#define CH     256                    // scan chunk length
#define NCH    (LSEQ / CH)            // 64 chunks

typedef __attribute__((ext_vector_type(16))) _Float16 v16h;
typedef __attribute__((ext_vector_type(8)))  float    v8f;
typedef unsigned int u32x4 __attribute__((ext_vector_type(4)));
typedef int          i32x4 __attribute__((ext_vector_type(4)));
typedef int          i32x8 __attribute__((ext_vector_type(8)));

__device__ __forceinline__ float warp_sum32(float v) {
#pragma unroll
  for (int m = 16; m >= 1; m >>= 1) v += __shfl_xor(v, m, 32);
  return v;
}

__device__ __forceinline__ float silu_f(float x) {
  return x / (1.0f + __expf(-x));
}

// load 8 contiguous floats (two global_load_b128)
__device__ __forceinline__ void load8(const float* __restrict__ p, float* t) {
  float4 a = ((const float4*)p)[0];
  float4 b = ((const float4*)p)[1];
  t[0] = a.x; t[1] = a.y; t[2] = a.z; t[3] = a.w;
  t[4] = b.x; t[5] = b.y; t[6] = b.z; t[7] = b.w;
}

// ---------------- [B,C,L] -> [B,L,C] ----------------
__global__ void to_seq_kernel(const float* __restrict__ src, float* __restrict__ dst) {
  size_t idx = (size_t)blockIdx.x * blockDim.x + threadIdx.x;
  if (idx >= (size_t)ROWS * DIMC) return;
  size_t c = idx & 63;
  size_t l = (idx >> 6) % LSEQ;
  size_t b = idx / ((size_t)LSEQ * DIMC);
  dst[idx] = src[(b * DIMC + c) * LSEQ + l];
}

// ---------------- LayerNorm over last dim (64), wave per row ----------------
__global__ void ln64_kernel(const float* __restrict__ x, const float* __restrict__ g,
                            const float* __restrict__ b, float* __restrict__ y) {
  int wid  = (blockIdx.x * blockDim.x + threadIdx.x) >> 5;
  int lane = threadIdx.x & 31;
  if (wid >= ROWS) return;
  const float* xr = x + (size_t)wid * DIMC;
  float v0 = xr[lane], v1 = xr[lane + 32];
  float mu = warp_sum32(v0 + v1) * (1.0f / 64.0f);
  float d0 = v0 - mu, d1 = v1 - mu;
  float var = warp_sum32(d0 * d0 + d1 * d1) * (1.0f / 64.0f);
  float rs  = rsqrtf(var + 1e-5f);
  float* yr = y + (size_t)wid * DIMC;
  yr[lane]      = d0 * rs * g[lane]      + b[lane];
  yr[lane + 32] = d1 * rs * g[lane + 32] + b[lane + 32];
}

// ---------------- WMMA A-fragment loader ----------------
// A (16x32 f16), contiguous k order: lane half-group g, element e ->
//   K = k0 + 16g + e (contiguous 16 floats)
template <bool AVG>
__device__ __forceinline__ v16h load_a_frag(const float* __restrict__ b1,
                                            const float* __restrict__ b2, int g) {
  float t[16], u[16];
  load8(b1 + 16 * g, t);
  load8(b1 + 16 * g + 8, t + 8);
  if (AVG) {
    load8(b2 + 16 * g, u);
    load8(b2 + 16 * g + 8, u + 8);
  }
  v16h a;
#pragma unroll
  for (int e = 0; e < 16; ++e)
    a[e] = (_Float16)(AVG ? 0.5f * (t[e] + u[e]) : t[e]);
  return a;
}

// ---------------- WMMA GEMM:  Y[M,N] = X[M,64] @ W[N,64]^T (+bias)(+resid) ----------
// gridDim.y sweeps 64-wide N slices; each block TDM-stages its [64,64] f32 weight
// slice into LDS, converts once to an [n][k] f16 tile, then each wave
// owns a 16-row strip and 4 n-tiles (32 acc VGPRs) with single-v16h B fragments.
template <int N, bool AVG, bool BIAS, bool RESID>
__global__ void __launch_bounds__(256)
gemm64_wmma(const float* __restrict__ X, const float* __restrict__ X2,
            const float* __restrict__ W, const float* __restrict__ bias,
            const float* __restrict__ resid, float* __restrict__ Y, int Mtot) {
  __shared__ alignas(16) float    ldsWf[64 * 64];   // raw f32 slice from TDM (16 KB)
  __shared__ alignas(32) _Float16 ldsT[64 * 64];    // [n][k] f16 tile        (8 KB)
  const int n_base = (N > 64) ? ((int)blockIdx.y << 6) : 0;

  // --- TDM stage: wave 0 DMAs the [64,64] f32 weight slice into LDS ---
  if ((threadIdx.x >> 5) == 0) {
    unsigned long long ga = (unsigned long long)(const void*)(W + (size_t)n_base * 64);
    unsigned lds_off = (unsigned)(unsigned long long)(void*)ldsWf; // LDS byte offset (low bits)
    const unsigned T = 64u * 64u;                                  // elements (f32)
    u32x4 g0 = {};
    g0[0] = 1u;                                        // count=1 (valid user D#)
    g0[1] = lds_off;                                   // lds_addr
    g0[2] = (unsigned)(ga & 0xFFFFFFFFu);              // global_addr lo
    g0[3] = (unsigned)((ga >> 32) & 0x1FFFFFFu) | (2u << 30); // global_addr hi | type=2
    i32x8 g1 = {};
    g1[0] = (int)(2u << 16);                           // data_size=4B, no multicast
    g1[1] = (int)((T & 0xFFFFu) << 16);                // tensor_dim0 [15:0]
    g1[2] = (int)(((T >> 16) & 0xFFFFu) | (1u << 16)); // tensor_dim0 hi | tensor_dim1=1
    g1[3] = (int)((T & 0xFFFFu) << 16);                // tile_dim0 = T
    g1[4] = 1;                                         // tile_dim1 = 1
    g1[5] = (int)T;                                    // tensor_dim0_stride
    i32x4 g2 = {}, g3 = {};
#if defined(__clang_major__) && __clang_major__ >= 23
    i32x8 g1b = {};
    __builtin_amdgcn_tensor_load_to_lds(g0, g1, g2, g3, g1b, 0);
#else
    __builtin_amdgcn_tensor_load_to_lds(g0, g1, g2, g3, 0);
#endif
    __builtin_amdgcn_s_wait_tensorcnt(0);
  }
  __syncthreads();

  // --- one-time f16 convert: ldsT[n*64 + k] = (f16) W[n][k] ---
  for (int idx = threadIdx.x; idx < 64 * 64; idx += 256) {
    int n = idx >> 6, k = idx & 63;
    ldsT[n * 64 + k] = (_Float16)ldsWf[idx];
  }
  __syncthreads();

  int wv = threadIdx.x >> 5, lane = threadIdx.x & 31;
  int mt = blockIdx.x * 8 + wv;
  if (mt >= (Mtot >> 4)) return;
  int m0 = mt << 4;
  int g  = lane >> 4;
  int mr = lane & 15;
  const float* arow  = X + (size_t)(m0 + mr) * 64;
  const float* arow2 = AVG ? (X2 + (size_t)(m0 + mr) * 64) : X;

  v8f zero = {};
  v8f acc[4];
#pragma unroll
  for (int nt = 0; nt < 4; ++nt) acc[nt] = zero;

#pragma unroll
  for (int k0 = 0; k0 < 64; k0 += 32) {
    v16h a = load_a_frag<AVG>(arow + k0, arow2 + k0, g);
    // B fragment: lane -> N, element -> K = k0 + 16g + e; one aligned 32B v16h LDS read per tile
    const v16h* brow = (const v16h*)(ldsT + mr * 64 + k0 + 16 * g);
#pragma unroll
    for (int nt = 0; nt < 4; ++nt) {
      acc[nt] = __builtin_amdgcn_wmma_f32_16x16x32_f16(false, a, false, brow[nt * 64],
                                                       (short)0, acc[nt], false, false);
    }
  }

  // C/D layout: VGPR r -> M = m0 + r + 8*(lane>=16), N = n_base + 16*nt + (lane&15)
#pragma unroll
  for (int nt = 0; nt < 4; ++nt) {
    int n = n_base + nt * 16 + mr;
    float bv = BIAS ? bias[n] : 0.0f;
#pragma unroll
    for (int r = 0; r < 8; ++r) {
      int m = m0 + r + (g << 3);
      float v = acc[nt][r] + bv;
      if (RESID) v += resid[(size_t)m * N + n];
      Y[(size_t)m * N + n] = v;
    }
  }
}

// ---------------- depthwise causal conv (K=4) + bias + SiLU (branchless pad) --------
__global__ void conv_silu_kernel(const float* __restrict__ x, int ldx,
                                 const float* __restrict__ w, const float* __restrict__ bias,
                                 float* __restrict__ y, int rev) {
  size_t idx = (size_t)blockIdx.x * blockDim.x + threadIdx.x;
  if (idx >= (size_t)ROWS * DIMC) return;
  int d = (int)(idx & 63);
  size_t row = idx >> 6;
  int l = (int)(row % LSEQ);
  int b = (int)(row / LSEQ);
  float acc = bias[d];
#pragma unroll
  for (int k = 0; k < DCONV; ++k) {
    int pos = l - (DCONV - 1) + k;
    int p   = pos < 0 ? 0 : pos;                 // clamp; mask weight instead of load
    int src = rev ? (LSEQ - 1 - p) : p;
    float wv = pos < 0 ? 0.0f : w[d * DCONV + k];
    acc += wv * x[((size_t)b * LSEQ + src) * ldx + d];
  }
  y[idx] = silu_f(acc);
}

// ---------------- x_proj (N=20) + dt_proj + softplus, wave per row ----------------
__global__ void xproj_dt_kernel(const float* __restrict__ sel,
                                const float* __restrict__ xw,   // [20,64]
                                const float* __restrict__ dtw,  // [64,4]
                                const float* __restrict__ dtb,  // [64]
                                float* __restrict__ dt, float* __restrict__ Bm,
                                float* __restrict__ Cm) {
  int wid  = (blockIdx.x * blockDim.x + threadIdx.x) >> 5;
  int lane = threadIdx.x & 31;
  if (wid >= ROWS) return;
  const float* s = sel + (size_t)wid * DIMC;
  float dv = 0.0f;
  if (lane < 20) {
    const float4* s4 = (const float4*)s;
    const float4* w4 = (const float4*)(xw + lane * DIMC);
#pragma unroll 4
    for (int j = 0; j < 16; ++j) {
      float4 a = s4[j], b = w4[j];
      dv += a.x * b.x + a.y * b.y + a.z * b.z + a.w * b.w;
    }
  }
  float r0 = __shfl(dv, 0, 32), r1 = __shfl(dv, 1, 32);
  float r2 = __shfl(dv, 2, 32), r3 = __shfl(dv, 3, 32);
  if (lane >= 4  && lane < 12) Bm[(size_t)wid * NSTATE + (lane - 4)]  = dv;
  if (lane >= 12 && lane < 20) Cm[(size_t)wid * NSTATE + (lane - 12)] = dv;
#pragma unroll
  for (int h = 0; h < 2; ++h) {
    int d = lane + 32 * h;
    float t = r0 * dtw[d * 4 + 0] + r1 * dtw[d * 4 + 1] +
              r2 * dtw[d * 4 + 2] + r3 * dtw[d * 4 + 3] + dtb[d];
    float sp = (t > 20.0f) ? t : __logf(1.0f + __expf(t));
    dt[(size_t)wid * DIMC + d] = sp;
  }
}

// ---------------- chunked linear-recurrence scan ----------------
// phase 1: per (b,d,chunk) reduce chunk to affine (A_prod, B_acc) per n
__global__ void scan_reduce_kernel(const float* __restrict__ dt, const float* __restrict__ xc,
                                   const float* __restrict__ Bm, const float* __restrict__ A_log,
                                   float* __restrict__ aggA, float* __restrict__ aggB) {
  int tid = blockIdx.x * blockDim.x + threadIdx.x;
  if (tid >= BATCH * DIMC * NCH) return;
  int d = tid & 63, b = (tid >> 6) & 7, c = tid >> 9;
  float a[NSTATE], ap[NSTATE], bb[NSTATE];
#pragma unroll
  for (int n = 0; n < NSTATE; ++n) {
    a[n] = -__expf(A_log[d * NSTATE + n]);
    ap[n] = 1.0f; bb[n] = 0.0f;
  }
  size_t base = (size_t)b * LSEQ + (size_t)c * CH;
  for (int i = 0; i < CH; ++i) {
    size_t row = base + i;
    float dtv = dt[row * DIMC + d];
    float xv  = xc[row * DIMC + d];
    float cm  = dtv * xv;
    float bm[NSTATE];
    load8(Bm + row * NSTATE, bm);          // two global_load_b128
#pragma unroll
    for (int n = 0; n < NSTATE; ++n) {
      float dA = __expf(dtv * a[n]);
      bb[n] = dA * bb[n] + cm * bm[n];
      ap[n] *= dA;
    }
  }
  int bd = b * DIMC + d;
#pragma unroll
  for (int n = 0; n < NSTATE; ++n) {
    aggA[((size_t)bd * NSTATE + n) * NCH + c] = ap[n];
    aggB[((size_t)bd * NSTATE + n) * NCH + c] = bb[n];
  }
}

// phase 2: tiny serial scan over chunk aggregates -> per-chunk initial state
__global__ void scan_combine_kernel(const float* __restrict__ aggA,
                                    const float* __restrict__ aggB,
                                    float* __restrict__ hinit) {
  int tid = blockIdx.x * blockDim.x + threadIdx.x;
  if (tid >= BATCH * DIMC) return;
#pragma unroll
  for (int n = 0; n < NSTATE; ++n) {
    size_t base = ((size_t)tid * NSTATE + n) * NCH;
    float h = 0.0f;
    for (int c = 0; c < NCH; ++c) {
      hinit[base + c] = h;
      h = aggA[base + c] * h + aggB[base + c];
    }
  }
}

// phase 3: re-apply recurrence with correct initial state, emit y = <h,C> + x*D
__global__ void scan_apply_kernel(const float* __restrict__ dt, const float* __restrict__ xc,
                                  const float* __restrict__ Bm, const float* __restrict__ Cm,
                                  const float* __restrict__ A_log, const float* __restrict__ Dp,
                                  const float* __restrict__ hinit, float* __restrict__ y) {
  int tid = blockIdx.x * blockDim.x + threadIdx.x;
  if (tid >= BATCH * DIMC * NCH) return;
  int d = tid & 63, b = (tid >> 6) & 7, c = tid >> 9;
  int bd = b * DIMC + d;
  float a[NSTATE], h[NSTATE];
#pragma unroll
  for (int n = 0; n < NSTATE; ++n) {
    a[n] = -__expf(A_log[d * NSTATE + n]);
    h[n] = hinit[((size_t)bd * NSTATE + n) * NCH + c];
  }
  float Dd = Dp[d];
  size_t base = (size_t)b * LSEQ + (size_t)c * CH;
  for (int i = 0; i < CH; ++i) {
    size_t row = base + i;
    if (i + 8 < CH) {  // gfx1250 global_prefetch_b8 for upcoming lines
      __builtin_prefetch(dt + (row + 8) * DIMC + d, 0, 1);
      __builtin_prefetch(xc + (row + 8) * DIMC + d, 0, 1);
    }
    float dtv = dt[row * DIMC + d];
    float xv  = xc[row * DIMC + d];
    float cm  = dtv * xv;
    float bm[NSTATE], cmv[NSTATE];
    load8(Bm + row * NSTATE, bm);
    load8(Cm + row * NSTATE, cmv);
    float acc = 0.0f;
#pragma unroll
    for (int n = 0; n < NSTATE; ++n) {
      float dA = __expf(dtv * a[n]);
      h[n] = dA * h[n] + cm * bm[n];
      acc += h[n] * cmv[n];
    }
    y[row * DIMC + d] = acc + xv * Dd;
  }
}

// ---------------- (yf [+ flip(yb)])(/2) * silu(z) then LayerNorm ----------------
template <bool BIDIR>
__global__ void gate_ln_kernel(const float* __restrict__ yf, const float* __restrict__ yb,
                               const float* __restrict__ z, int ldz,
                               const float* __restrict__ gw, const float* __restrict__ gb,
                               float* __restrict__ out) {
  int wid  = (blockIdx.x * blockDim.x + threadIdx.x) >> 5;
  int lane = threadIdx.x & 31;
  if (wid >= ROWS) return;
  int l = wid % LSEQ, b = wid / LSEQ;
  size_t rrow = (size_t)b * LSEQ + (LSEQ - 1 - l);
  float v[2];
#pragma unroll
  for (int h = 0; h < 2; ++h) {
    int d = lane + 32 * h;
    float yv = yf[(size_t)wid * DIMC + d];
    if (BIDIR) yv = 0.5f * (yv + yb[rrow * DIMC + d]);
    float zv = z[(size_t)wid * ldz + d];
    v[h] = yv * silu_f(zv);
  }
  float mu = warp_sum32(v[0] + v[1]) * (1.0f / 64.0f);
  float d0 = v[0] - mu, d1 = v[1] - mu;
  float var = warp_sum32(d0 * d0 + d1 * d1) * (1.0f / 64.0f);
  float rs  = rsqrtf(var + 1e-5f);
  out[(size_t)wid * DIMC + lane]      = d0 * rs * gw[lane]      + gb[lane];
  out[(size_t)wid * DIMC + lane + 32] = d1 * rs * gw[lane + 32] + gb[lane + 32];
}

// ---------------- final transpose/add: out0=[B,C,L] pan_s ; out1 = ms_s + fus ----------
__global__ void finalize_kernel(const float* __restrict__ pan_s, const float* __restrict__ ms_s,
                                const float* __restrict__ fus, float* __restrict__ out) {
  size_t idx = (size_t)blockIdx.x * blockDim.x + threadIdx.x;
  size_t total = (size_t)ROWS * DIMC;
  if (idx >= total) return;
  size_t l = idx % LSEQ;
  size_t c = (idx / LSEQ) % DIMC;
  size_t b = idx / ((size_t)LSEQ * DIMC);
  size_t src = (b * LSEQ + l) * DIMC + c;
  out[idx]         = pan_s[src];
  out[total + idx] = ms_s[src] + fus[src];
}

// ======================= host-side orchestration =======================
namespace {

struct Ws {
  float *pan_s, *ms_s, *spa_f, *spe_f;
  float *xn, *extra_n, *xz, *xc, *dt, *yf, *yb;
  float *Bm, *Cm, *aggA, *aggB, *hinit;
};

struct V6P {
  const float *ln_g, *ln_b, *in_w, *conv_w, *conv_b, *x_proj_w, *dt_w, *dt_b,
              *A_log, *D, *norm_g, *norm_b, *out_w, *out_b,
              *conv_w_b, *conv_b_b, *A_log_b, *D_b;
};
struct V7P {
  const float *ln0_g, *ln0_b, *ln1_g, *ln1_b,
              *in_w, *conv_w, *conv_b, *x_proj_w, *dt_w, *dt_b,
              *A_log, *D, *norm_g, *norm_b, *out_w, *out_b;
};

constexpr int GEMM_BLOCKS = (ROWS / 16) / 8;   // 1024: 8 waves/block, 16 rows/wave

inline void run_scan(hipStream_t s, const Ws& w, const float* xc,
                     const float* A_log, const float* Dp, float* y) {
  scan_reduce_kernel<<<(BATCH * DIMC * NCH + 255) / 256, 256, 0, s>>>(
      w.dt, xc, w.Bm, A_log, w.aggA, w.aggB);
  scan_combine_kernel<<<(BATCH * DIMC + 255) / 256, 256, 0, s>>>(w.aggA, w.aggB, w.hinit);
  scan_apply_kernel<<<(BATCH * DIMC * NCH + 255) / 256, 256, 0, s>>>(
      w.dt, xc, w.Bm, w.Cm, A_log, Dp, w.hinit, y);
}

inline void run_v6(hipStream_t s, const Ws& w, const V6P& p, float* state) {
  const int rowsB = ROWS / 8;
  const int elB   = (ROWS * DIMC + 255) / 256;
  ln64_kernel<<<rowsB, 256, 0, s>>>(state, p.ln_g, p.ln_b, w.xn);
  gemm64_wmma<128, false, false, false><<<dim3(GEMM_BLOCKS, 2), 256, 0, s>>>(
      w.xn, nullptr, p.in_w, nullptr, nullptr, w.xz, ROWS);
  // forward branch
  conv_silu_kernel<<<elB, 256, 0, s>>>(w.xz, 128, p.conv_w, p.conv_b, w.xc, 0);
  xproj_dt_kernel<<<rowsB, 256, 0, s>>>(w.xc, p.x_proj_w, p.dt_w, p.dt_b, w.dt, w.Bm, w.Cm);
  run_scan(s, w, w.xc, p.A_log, p.D, w.yf);
  // backward branch (reversed-index domain throughout)
  conv_silu_kernel<<<elB, 256, 0, s>>>(w.xz, 128, p.conv_w_b, p.conv_b_b, w.xc, 1);
  xproj_dt_kernel<<<rowsB, 256, 0, s>>>(w.xc, p.x_proj_w, p.dt_w, p.dt_b, w.dt, w.Bm, w.Cm);
  run_scan(s, w, w.xc, p.A_log_b, p.D_b, w.yb);
  // gate + norm + out_proj + residual (in place on state)
  gate_ln_kernel<true><<<rowsB, 256, 0, s>>>(w.yf, w.yb, w.xz + 64, 128,
                                             p.norm_g, p.norm_b, w.xn);
  gemm64_wmma<64, false, true, true><<<dim3(GEMM_BLOCKS, 1), 256, 0, s>>>(
      w.xn, nullptr, p.out_w, p.out_b, state, state, ROWS);
}

inline void run_v7(hipStream_t s, const Ws& w, const V7P& p,
                   const float* state_u, const float* extra, float* dst) {
  const int rowsB = ROWS / 8;
  const int elB   = (ROWS * DIMC + 255) / 256;
  ln64_kernel<<<rowsB, 256, 0, s>>>(state_u, p.ln0_g, p.ln0_b, w.xn);
  ln64_kernel<<<rowsB, 256, 0, s>>>(extra,   p.ln1_g, p.ln1_b, w.extra_n);
  gemm64_wmma<128, false, false, false><<<dim3(GEMM_BLOCKS, 2), 256, 0, s>>>(
      w.xn, nullptr, p.in_w, nullptr, nullptr, w.xz, ROWS);
  conv_silu_kernel<<<elB, 256, 0, s>>>(w.xz, 128, p.conv_w, p.conv_b, w.xc, 0);
  xproj_dt_kernel<<<rowsB, 256, 0, s>>>(w.extra_n, p.x_proj_w, p.dt_w, p.dt_b,
                                        w.dt, w.Bm, w.Cm);
  run_scan(s, w, w.xc, p.A_log, p.D, w.yf);
  gate_ln_kernel<false><<<rowsB, 256, 0, s>>>(w.yf, nullptr, w.xz + 64, 128,
                                              p.norm_g, p.norm_b, w.xn);
  gemm64_wmma<64, false, true, true><<<dim3(GEMM_BLOCKS, 1), 256, 0, s>>>(
      w.xn, nullptr, p.out_w, p.out_b, state_u, dst, ROWS);
}

} // namespace

extern "C" void kernel_launch(void* const* d_in, const int* in_sizes, int n_in,
                              void* d_out, int out_size, void* d_ws, size_t ws_size,
                              hipStream_t stream) {
  (void)in_sizes; (void)n_in; (void)out_size; (void)ws_size;
  auto F = [&](int i) { return (const float*)d_in[i]; };

  // ---- parameter unpacking (setup_inputs dict order, recursive) ----
  // 0=pan 1=ms ; spa0 @2 ; spe0 @20 ; cross_spa @38 ; cross_spe @54 ; out_w=70 out_b=71
  auto v6at = [&](int o) {
    V6P p{F(o+0),F(o+1),F(o+2),F(o+3),F(o+4),F(o+5),F(o+6),F(o+7),F(o+8),F(o+9),
          F(o+10),F(o+11),F(o+12),F(o+13),F(o+14),F(o+15),F(o+16),F(o+17)};
    return p;
  };
  auto v7at = [&](int o) {
    V7P p{F(o+0),F(o+1),F(o+2),F(o+3),F(o+4),F(o+5),F(o+6),F(o+7),F(o+8),F(o+9),
          F(o+10),F(o+11),F(o+12),F(o+13),F(o+14),F(o+15)};
    return p;
  };
  V6P spa = v6at(2), spe = v6at(20);
  V7P cspa = v7at(38), cspe = v7at(54);
  const float* fus_w = F(70);
  const float* fus_b = F(71);

  // ---- workspace layout (floats). S = 8.39M floats per [B,L,64] buffer ----
  const size_t S = (size_t)ROWS * DIMC;
  float* wsf = (float*)d_ws;
  Ws w;
  w.pan_s   = wsf + 0 * S;
  w.ms_s    = wsf + 1 * S;
  w.spa_f   = wsf + 2 * S;
  w.spe_f   = wsf + 3 * S;
  w.xn      = wsf + 4 * S;
  w.extra_n = wsf + 5 * S;
  w.xz      = wsf + 6 * S;          // 2*S
  w.xc      = wsf + 8 * S;
  w.dt      = wsf + 9 * S;
  w.yf      = wsf + 10 * S;
  w.yb      = wsf + 11 * S;
  float* small = wsf + 12 * S;
  w.Bm    = small;                  small += (size_t)ROWS * NSTATE;
  w.Cm    = small;                  small += (size_t)ROWS * NSTATE;
  w.aggA  = small;                  small += (size_t)BATCH * DIMC * NSTATE * NCH;
  w.aggB  = small;                  small += (size_t)BATCH * DIMC * NSTATE * NCH;
  w.hinit = small;

  const int elB = (int)((S + 255) / 256);

  // ---- [B,C,H,W] -> [B,L,C] sequences ----
  to_seq_kernel<<<elB, 256, 0, stream>>>(F(0), w.pan_s);
  to_seq_kernel<<<elB, 256, 0, stream>>>(F(1), w.ms_s);

  // ---- DEPTH=1 bidirectional blocks ----
  run_v6(stream, w, spa, w.pan_s);
  run_v6(stream, w, spe, w.ms_s);

  // ---- cross blocks ----
  run_v7(stream, w, cspa, w.pan_s, w.ms_s, w.spa_f);
  run_v7(stream, w, cspe, w.ms_s, w.pan_s, w.spe_f);

  // ---- fusion GEMM: ((spa_f + spe_f)/2) @ out_w^T + out_b  (avg folded into A load) ----
  float* fus = w.xz; // xz scratch is free now
  gemm64_wmma<64, true, true, false><<<dim3(GEMM_BLOCKS, 1), 256, 0, stream>>>(
      w.spa_f, w.spe_f, fus_w, fus_b, nullptr, fus, ROWS);

  // ---- outputs: pan_out ; ms_out + fusion (both [B,C,H,W]) ----
  finalize_kernel<<<elB, 256, 0, stream>>>(w.pan_s, w.ms_s, fus, (float*)d_out);
}
